// ConfigurableGraphMoE_72267119722661
// MI455X (gfx1250) — matrix operand order repaired and checked by reading, not compiled, not run on hardware
//
#include <hip/hip_runtime.h>

// ---------------- problem constants (from reference) ----------------
#define NN 50000
#define EE 800000
#define HH 256
#define OO 128
#define GG 64

typedef __attribute__((ext_vector_type(16))) __bf16          v16bf;
typedef __attribute__((ext_vector_type(16))) unsigned short  v16u;
typedef __attribute__((ext_vector_type(8)))  unsigned short  v8u;
typedef __attribute__((ext_vector_type(8)))  float           v8f;

__device__ __forceinline__ unsigned short f2bf(float f) {
  unsigned int u = __float_as_uint(f);
  unsigned int r = u + 0x7fffu + ((u >> 16) & 1u);   // round-to-nearest-even
  return (unsigned short)(r >> 16);
}
__device__ __forceinline__ float bf2f(unsigned short h) {
  return __uint_as_float(((unsigned int)h) << 16);
}

// gfx1250 async copy: global -> LDS, 128-bit per lane, tracked by ASYNCcnt.
__device__ __forceinline__ void async_ld_b128(unsigned lds_off, const void* gaddr) {
  asm volatile("global_load_async_to_lds_b128 %0, %1, off"
               :: "v"(lds_off), "v"((unsigned long long)(size_t)gaddr)
               : "memory");
}

// ---------------- tiny utility kernels ----------------
__global__ void zero_f32_kernel(float* __restrict__ p, long n) {
  long i = (long)blockIdx.x * blockDim.x + threadIdx.x;
  if (i < n) p[i] = 0.0f;
}

__global__ void cvt_f32_bf16_kernel(const float* __restrict__ in,
                                    unsigned short* __restrict__ out, long n) {
  long i = (long)blockIdx.x * blockDim.x + threadIdx.x;
  if (i < n) out[i] = f2bf(in[i]);
}

// weights: fp32 row-major [z][K][Oc] -> bf16 transposed [z][Oc][K]
__global__ void cvt_t_kernel(const float* __restrict__ in, unsigned short* __restrict__ out,
                             int K, int Oc, long total) {
  long idx = (long)blockIdx.x * blockDim.x + threadIdx.x;
  if (idx >= total) return;
  long zo = idx / K;
  int  k  = (int)(idx % K);
  long z  = zo / Oc;
  int  o  = (int)(zo % Oc);
  out[idx] = f2bf(in[z * (long)K * Oc + (long)k * Oc + o]);
}

// ---------------- graph stats ----------------
__global__ void node_count_kernel(const int* __restrict__ batch, float* __restrict__ cn) {
  long i = (long)blockIdx.x * blockDim.x + threadIdx.x;
  if (i < NN) atomicAdd(&cn[batch[i]], 1.0f);
}
__global__ void edge_count_kernel(const int* __restrict__ src, const int* __restrict__ batch,
                                  float* __restrict__ ce) {
  long i = (long)blockIdx.x * blockDim.x + threadIdx.x;
  if (i < EE) atomicAdd(&ce[batch[src[i]]], 1.0f);
}

__global__ void finalize_stats_kernel(const float* __restrict__ cn, const float* __restrict__ ce,
                                      float* __restrict__ featsN, float* __restrict__ lognN) {
  __shared__ float sa[GG], sb[GG];
  int g = threadIdx.x;                       // 64 threads
  float n    = fmaxf(cn[g], 1.0f);
  float e    = ce[g];
  float dens = e / fmaxf(n * (n - 1.0f), 1.0f);
  float lgn  = logf(n);
  sa[g] = lgn; sb[g] = lgn; __syncthreads();
  for (int s = 32; s > 0; s >>= 1) {
    if (g < s) { sa[g] = fminf(sa[g], sa[g + s]); sb[g] = fmaxf(sb[g], sb[g + s]); }
    __syncthreads();
  }
  float lmin = sa[0], lmax = sb[0]; __syncthreads();
  lognN[g] = (lgn - lmin) / (lmax - lmin + 1e-6f);
  float f[3] = { n, e, dens };
  for (int c = 0; c < 3; ++c) {
    sa[g] = f[c]; __syncthreads();
    for (int s = 32; s > 0; s >>= 1) { if (g < s) sa[g] += sa[g + s]; __syncthreads(); }
    float mean = sa[0] / (float)GG; __syncthreads();
    float d = f[c] - mean;
    sa[g] = d * d; __syncthreads();
    for (int s = 32; s > 0; s >>= 1) { if (g < s) sa[g] += sa[g + s]; __syncthreads(); }
    float stdv = sqrtf(sa[0] / (float)GG); __syncthreads();
    featsN[g * 3 + c] = d / (stdv + 1e-6f);
  }
}

// ---------------- encoder layer 1 (K=6, scalar) ----------------
__global__ void encoder1_kernel(const float* __restrict__ x, const float* __restrict__ W1,
                                const float* __restrict__ b1, unsigned short* __restrict__ out) {
  long idx = (long)blockIdx.x * blockDim.x + threadIdx.x;
  if (idx >= (long)NN * HH) return;
  int n = (int)(idx >> 8), o = (int)(idx & 255);
  float a = b1[o];
#pragma unroll
  for (int k = 0; k < 6; ++k) a += x[(size_t)n * 16 + 4 + k] * W1[k * HH + o];
  out[idx] = f2bf(fmaxf(a, 0.0f));
}

// ---------------- edge aggregation: out[dst] += bf16(v[src]) ----------------
__global__ void aggregate_kernel(const int* __restrict__ src, const int* __restrict__ dst,
                                 const unsigned short* __restrict__ v, float* __restrict__ out,
                                 long total) {
  long idx = (long)blockIdx.x * blockDim.x + threadIdx.x;
  if (idx >= total) return;
  long e = idx >> 6;                 // HH/4 = 64 chunks of 4 per edge
  int  c = (int)(idx & 63) << 2;
  int  s = src[e], d = dst[e];
  uint2 u = *(const uint2*)(v + (size_t)s * HH + c);
  float* po = out + (size_t)d * HH + c;
  atomicAdd(po + 0, bf2f((unsigned short)(u.x & 0xffff)));
  atomicAdd(po + 1, bf2f((unsigned short)(u.x >> 16)));
  atomicAdd(po + 2, bf2f((unsigned short)(u.y & 0xffff)));
  atomicAdd(po + 3, bf2f((unsigned short)(u.y >> 16)));
}

// ---------------- router: LN + relu + 256->4 head + softmax + top2 ----------------
__global__ __launch_bounds__(256)
void router_kernel(const float* __restrict__ t, const float* __restrict__ ln_g,
                   const float* __restrict__ ln_b, const float* __restrict__ r_W2,
                   const float* __restrict__ r_b2, const float* __restrict__ centers,
                   const int* __restrict__ batch, const float* __restrict__ lognN,
                   float* __restrict__ wts) {
  int wave = threadIdx.x >> 5, lane = threadIdx.x & 31;
  int n = blockIdx.x * 8 + wave;
  if (n >= NN) return;
  const float* tn = t + (size_t)n * HH;
  float v[8], s = 0.0f;
#pragma unroll
  for (int j = 0; j < 8; ++j) { v[j] = tn[lane + 32 * j]; s += v[j]; }
  for (int off = 16; off > 0; off >>= 1) s += __shfl_xor(s, off, 32);
  float mean = s * (1.0f / HH), s2 = 0.0f;
#pragma unroll
  for (int j = 0; j < 8; ++j) { float d = v[j] - mean; s2 += d * d; }
  for (int off = 16; off > 0; off >>= 1) s2 += __shfl_xor(s2, off, 32);
  float rstd = rsqrtf(s2 * (1.0f / HH) + 1e-5f);
  float acc[4] = {0, 0, 0, 0};
#pragma unroll
  for (int j = 0; j < 8; ++j) {
    int k = lane + 32 * j;
    float u = fmaxf((v[j] - mean) * rstd * ln_g[k] + ln_b[k], 0.0f);
#pragma unroll
    for (int c = 0; c < 4; ++c) acc[c] += u * r_W2[k * 4 + c];
  }
#pragma unroll
  for (int c = 0; c < 4; ++c)
    for (int off = 16; off > 0; off >>= 1) acc[c] += __shfl_xor(acc[c], off, 32);
  if (lane == 0) {
    float lnn = lognN[batch[n]];
    float logits[4], lmax = -1e30f;
#pragma unroll
    for (int c = 0; c < 4; ++c) {
      float d = lnn - centers[c];
      logits[c] = 0.7f * (acc[c] + r_b2[c]) - 0.3f * d * d;
      lmax = fmaxf(lmax, logits[c]);
    }
    float pr[4], se = 0.0f;
#pragma unroll
    for (int c = 0; c < 4; ++c) { pr[c] = expf(logits[c] - lmax); se += pr[c]; }
#pragma unroll
    for (int c = 0; c < 4; ++c) pr[c] /= se;
    int i1 = 0;
#pragma unroll
    for (int c = 1; c < 4; ++c) if (pr[c] > pr[i1]) i1 = c;
    int i2 = -1;
#pragma unroll
    for (int c = 0; c < 4; ++c) if (c != i1 && (i2 < 0 || pr[c] > pr[i2])) i2 = c;
    float tv = pr[i1] + pr[i2] + 1e-8f;
    float* w = wts + (size_t)n * 4;
#pragma unroll
    for (int c = 0; c < 4; ++c) w[c] = 0.0f;
    w[i1] = pr[i1] / tv;
    w[i2] = pr[i2] / tv;
  }
}

// ---------------- WMMA dual-input GEMM, async double-buffered ----------------
// D[M x Oc] = epilogue( A1@B1t + (A2@B2t) + bias ), bf16 in, f32 accum.
// A row-major [M][K]; B pre-transposed [Oc][K]. Block tile 64x128, 8 waves
// (4 M-subtiles x 2 N-groups of 64); each wave holds one A fragment reused by
// 4 WMMAs. K step 32, 2-deep LDS pipeline fed by global_load_async_to_lds_b128
// (ASYNCcnt): 3 async copies/thread/stage, steady-state s_wait_asynccnt 0x3.
// MODE 0: (+relu) -> bf16 out   MODE 1: + sf@rW1_tail -> f32 (router t)
// MODE 2: d_out[n,:] += wts[n,expert] * val
template<int MODE, bool RELU>
__global__ __launch_bounds__(256)
void gemm_dual(const unsigned short* __restrict__ A1, const unsigned short* __restrict__ B1base,
               const unsigned short* __restrict__ A2, const unsigned short* __restrict__ B2base,
               const float* __restrict__ biasBase, int M, int K, int Oc,
               long strideB, long strideBias, long strideOut,
               unsigned short* __restrict__ outBfBase, float* __restrict__ outF,
               const int* __restrict__ batch, const float* __restrict__ feats,
               const float* __restrict__ w1tail, const float* __restrict__ wts, int expert) {
  __shared__ __align__(32) unsigned short As[2][64][32];
  __shared__ __align__(32) unsigned short Bs[2][128][32];

  const int z = blockIdx.z;
  const unsigned short* B1 = B1base + (long)z * strideB;
  const unsigned short* B2 = B2base ? B2base + (long)z * strideB : nullptr;
  const float* bias = biasBase + (long)z * strideBias;
  unsigned short* outBf = outBfBase ? outBfBase + (long)z * strideOut : nullptr;

  const int tid  = threadIdx.x;
  const int lane = tid & 31, wave = tid >> 5;
  const int wm = wave & 3, wn = wave >> 2;
  const int rowBlock = blockIdx.x * 64;
  const int colBlock = blockIdx.y * 128;

  v8f acc[4];
#pragma unroll
  for (int q = 0; q < 4; ++q) acc[q] = (v8f){0, 0, 0, 0, 0, 0, 0, 0};

  // tile loaders: A tile 64x32 (1 x 16B/thread), B tile 128x32 (2 x 16B/thread)
  const int l_row = tid >> 2;                  // 0..63
  const int l_kc  = (tid & 3) << 3;            // 0,8,16,24
  const int grc   = min(rowBlock + l_row, M - 1);  // clamp ragged M (rows >= M never stored)

  const int fr    = lane & 15;
  const int kbase = (lane < 16) ? 0 : 16;      // A frag K half (contiguous, matches B)
  const int koff  = (lane < 16) ? 0 : 16;      // B frag K half

  const int nk = K >> 5;
  const int nst = ((A2 != nullptr && B2 != nullptr) ? 2 : 1) * nk;

  auto issueStage = [&](int st, int buf) {
    const unsigned short* A = (st >= nk) ? A2 : A1;
    const unsigned short* B = (st >= nk) ? B2 : B1;
    const int kk = ((st >= nk) ? (st - nk) : st) << 5;
    async_ld_b128((unsigned)(size_t)&As[buf][l_row][l_kc],
                  A + (size_t)grc * K + kk + l_kc);
    async_ld_b128((unsigned)(size_t)&Bs[buf][l_row][l_kc],
                  B + (size_t)(colBlock + l_row) * K + kk + l_kc);
    async_ld_b128((unsigned)(size_t)&Bs[buf][l_row + 64][l_kc],
                  B + (size_t)(colBlock + l_row + 64) * K + kk + l_kc);
  };

  issueStage(0, 0);
  for (int st = 0; st < nst; ++st) {
    const int buf = st & 1;
    if (st + 1 < nst) {
      issueStage(st + 1, buf ^ 1);
      asm volatile("s_wait_asynccnt 0x3" ::: "memory");   // current stage's 3 copies done
    } else {
      asm volatile("s_wait_asynccnt 0x0" ::: "memory");
    }
    __syncthreads();

    // A fragment: lane<16 -> K {0..15}; lane>=16 -> K {16..31}
    v8u alo = *(const v8u*)&As[buf][wm * 16 + fr][kbase];
    v8u ahi = *(const v8u*)&As[buf][wm * 16 + fr][kbase + 8];
    v16u au;
#pragma unroll
    for (int j = 0; j < 8; ++j) { au[j] = alo[j]; au[j + 8] = ahi[j]; }
    v16bf af = __builtin_bit_cast(v16bf, au);
    // 4 B fragments share the A fragment: cols wn*64 + {0,16,32,48} + fr
#pragma unroll
    for (int q = 0; q < 4; ++q) {
      v16u bu = *(const v16u*)&Bs[buf][wn * 64 + q * 16 + fr][koff];
      v16bf bfq = __builtin_bit_cast(v16bf, bu);
      acc[q] = __builtin_amdgcn_wmma_f32_16x16x32_bf16(false, af, false, bfq, (short)0, acc[q], false, false);
    }
    __syncthreads();
  }

  // epilogue — C layout: lane<16: VGPR i -> M=i ; lane>=16: M=i+8 ; N = lane&15
  int rbase = rowBlock + wm * 16 + ((lane >> 4) << 3);
  int col[4];
  float bv[4];
#pragma unroll
  for (int q = 0; q < 4; ++q) { col[q] = colBlock + wn * 64 + q * 16 + fr; bv[q] = bias[col[q]]; }
#pragma unroll
  for (int i = 0; i < 8; ++i) {
    int r = rbase + i;
    if (r >= M) continue;
    if (MODE == 1) {
      int g = batch[r];
      float s0 = feats[g * 3 + 0], s1 = feats[g * 3 + 1], s2 = feats[g * 3 + 2];
#pragma unroll
      for (int q = 0; q < 4; ++q) {
        float v = acc[q][i] + bv[q]
                + s0 * w1tail[col[q]] + s1 * w1tail[Oc + col[q]] + s2 * w1tail[2 * Oc + col[q]];
        outF[(size_t)r * Oc + col[q]] = v;
      }
    } else if (MODE == 2) {
      float w = wts[(size_t)r * 4 + expert];
#pragma unroll
      for (int q = 0; q < 4; ++q)
        outF[(size_t)r * Oc + col[q]] += w * (acc[q][i] + bv[q]);
    } else {
#pragma unroll
      for (int q = 0; q < 4; ++q) {
        float v = acc[q][i] + bv[q];
        if (RELU) v = fmaxf(v, 0.0f);
        outBf[(size_t)r * Oc + col[q]] = f2bf(v);
      }
    }
  }
}

// ---------------- host orchestration ----------------
static inline long cdiv(long a, long b) { return (a + b - 1) / b; }

extern "C" void kernel_launch(void* const* d_in, const int* in_sizes, int n_in,
                              void* d_out, int out_size, void* d_ws, size_t ws_size,
                              hipStream_t stream) {
  (void)in_sizes; (void)n_in; (void)out_size; (void)ws_size;
  const float* x       = (const float*)d_in[0];
  const int*   eidx    = (const int*)d_in[1];
  const int*   batch   = (const int*)d_in[2];
  const float* enc_W1  = (const float*)d_in[3];
  const float* enc_b1  = (const float*)d_in[4];
  const float* enc_W2  = (const float*)d_in[5];
  const float* enc_b2  = (const float*)d_in[6];
  const float* r_W1    = (const float*)d_in[7];
  const float* r_b1    = (const float*)d_in[8];
  const float* ln_g    = (const float*)d_in[9];
  const float* ln_b    = (const float*)d_in[10];
  const float* r_W2    = (const float*)d_in[11];
  const float* r_b2    = (const float*)d_in[12];
  const float* centers = (const float*)d_in[13];
  const float* W0_rel  = (const float*)d_in[14];
  const float* b0_rel  = (const float*)d_in[15];
  const float* W0_root = (const float*)d_in[16];
  const float* W1_rel  = (const float*)d_in[17];
  const float* b1_rel  = (const float*)d_in[18];
  const float* W1_root = (const float*)d_in[19];
  const float* W2_rel  = (const float*)d_in[20];
  const float* b2_rel  = (const float*)d_in[21];
  const float* W2_root = (const float*)d_in[22];
  const int* src = eidx;
  const int* dst = eidx + EE;
  float* out = (float*)d_out;

  // workspace carve
  char* p = (char*)d_ws;
  auto carve = [&](size_t bytes) -> char* {
    char* r = p; p += (bytes + 255) & ~(size_t)255; return r;
  };
  unsigned short* encW2b = (unsigned short*)carve((size_t)HH * HH * 2);       // transposed
  unsigned short* rW1b   = (unsigned short*)carve((size_t)HH * HH * 2);       // transposed, rows 0..255
  unsigned short* W0rb   = (unsigned short*)carve((size_t)4 * HH * HH * 2);   // transposed
  unsigned short* W0tb   = (unsigned short*)carve((size_t)4 * HH * HH * 2);
  unsigned short* W1rb   = (unsigned short*)carve((size_t)4 * HH * HH * 2);
  unsigned short* W1tb   = (unsigned short*)carve((size_t)4 * HH * HH * 2);
  unsigned short* W2rb   = (unsigned short*)carve((size_t)4 * HH * OO * 2);
  unsigned short* W2tb   = (unsigned short*)carve((size_t)4 * HH * OO * 2);
  unsigned short* t1b    = (unsigned short*)carve((size_t)NN * HH * 2);
  unsigned short* hb     = (unsigned short*)carve((size_t)NN * HH * 2);
  unsigned short* aggb   = (unsigned short*)carve((size_t)NN * HH * 2);
  unsigned short* hEb    = (unsigned short*)carve((size_t)4 * NN * HH * 2);
  unsigned short* hE2b   = (unsigned short*)carve((size_t)4 * NN * HH * 2);
  float* scrF   = (float*)carve((size_t)NN * HH * 4);   // router t, then agg accumulator
  float* wts    = (float*)carve((size_t)NN * 4 * 4);
  float* cn     = (float*)carve(GG * 4);
  float* ce     = (float*)carve(GG * 4);
  float* featsN = (float*)carve(GG * 3 * 4);
  float* lognN  = (float*)carve(GG * 4);

  const int TB = 256;
  auto cvtT = [&](const float* in, unsigned short* o, int K, int Oc, int nz) {
    long total = (long)nz * K * Oc;
    cvt_t_kernel<<<(unsigned)cdiv(total, TB), TB, 0, stream>>>(in, o, K, Oc, total);
  };
  auto zero = [&](float* o, long n) {
    zero_f32_kernel<<<(unsigned)cdiv(n, TB), TB, 0, stream>>>(o, n);
  };

  // 1) weights -> bf16, transposed to [Oc][K]
  cvtT(enc_W2,  encW2b, HH, HH, 1);
  cvtT(r_W1,    rW1b,   HH, HH, 1);            // reads rows 0..255 of (259 x 256)
  cvtT(W0_rel,  W0rb,   HH, HH, 4);
  cvtT(W0_root, W0tb,   HH, HH, 4);
  cvtT(W1_rel,  W1rb,   HH, HH, 4);
  cvtT(W1_root, W1tb,   HH, HH, 4);
  cvtT(W2_rel,  W2rb,   HH, OO, 4);
  cvtT(W2_root, W2tb,   HH, OO, 4);

  // 2) graph stats
  zero(cn, GG); zero(ce, GG);
  node_count_kernel<<<(unsigned)cdiv(NN, TB), TB, 0, stream>>>(batch, cn);
  edge_count_kernel<<<(unsigned)cdiv(EE, TB), TB, 0, stream>>>(src, batch, ce);
  finalize_stats_kernel<<<1, GG, 0, stream>>>(cn, ce, featsN, lognN);

  // 3) encoder
  encoder1_kernel<<<(unsigned)cdiv((long)NN * HH, TB), TB, 0, stream>>>(x, enc_W1, enc_b1, t1b);
  const unsigned MT = (unsigned)cdiv(NN, 64);
  gemm_dual<0, false><<<dim3(MT, HH / 128, 1), TB, 0, stream>>>(
      t1b, encW2b, nullptr, nullptr, enc_b2, NN, HH, HH, 0, 0, 0,
      hb, nullptr, nullptr, nullptr, nullptr, nullptr, 0);

  // 4) router
  gemm_dual<1, false><<<dim3(MT, HH / 128, 1), TB, 0, stream>>>(
      hb, rW1b, nullptr, nullptr, r_b1, NN, HH, HH, 0, 0, 0,
      nullptr, scrF, batch, featsN, r_W1 + (size_t)HH * HH, nullptr, 0);
  router_kernel<<<(unsigned)cdiv(NN, 8), TB, 0, stream>>>(
      scrF, ln_g, ln_b, r_W2, r_b2, centers, batch, lognN, wts);

  // 5) hop 0: agg_h = aggregate(h); hE = relu(agg@W0_rel + h@W0_root + b0)
  zero(scrF, (long)NN * HH);
  aggregate_kernel<<<(unsigned)cdiv((long)EE * (HH / 4), TB), TB, 0, stream>>>(
      src, dst, hb, scrF, (long)EE * (HH / 4));
  cvt_f32_bf16_kernel<<<(unsigned)cdiv((long)NN * HH, TB), TB, 0, stream>>>(
      scrF, aggb, (long)NN * HH);
  gemm_dual<0, true><<<dim3(MT, HH / 128, 4), TB, 0, stream>>>(
      aggb, W0rb, hb, W0tb, b0_rel, NN, HH, HH,
      (long)HH * HH, HH, (long)NN * HH,
      hEb, nullptr, nullptr, nullptr, nullptr, nullptr, 0);

  // 6) hop 1 per expert
  for (int e = 0; e < 4; ++e) {
    zero(scrF, (long)NN * HH);
    aggregate_kernel<<<(unsigned)cdiv((long)EE * (HH / 4), TB), TB, 0, stream>>>(
        src, dst, hEb + (size_t)e * NN * HH, scrF, (long)EE * (HH / 4));
    cvt_f32_bf16_kernel<<<(unsigned)cdiv((long)NN * HH, TB), TB, 0, stream>>>(
        scrF, aggb, (long)NN * HH);
    gemm_dual<0, true><<<dim3(MT, HH / 128, 1), TB, 0, stream>>>(
        aggb, W1rb + (size_t)e * HH * HH, hEb + (size_t)e * NN * HH, W1tb + (size_t)e * HH * HH,
        b1_rel + (size_t)e * HH, NN, HH, HH, 0, 0, 0,
        hE2b + (size_t)e * NN * HH, nullptr, nullptr, nullptr, nullptr, nullptr, 0);
  }

  // 7) hop 2 per expert, weighted directly into d_out
  zero(out, (long)NN * OO);
  for (int e = 0; e < 4; ++e) {
    zero(scrF, (long)NN * HH);
    aggregate_kernel<<<(unsigned)cdiv((long)EE * (HH / 4), TB), TB, 0, stream>>>(
        src, dst, hE2b + (size_t)e * NN * HH, scrF, (long)EE * (HH / 4));
    cvt_f32_bf16_kernel<<<(unsigned)cdiv((long)NN * HH, TB), TB, 0, stream>>>(
        scrF, aggb, (long)NN * HH);
    gemm_dual<2, false><<<dim3(MT, OO / 128, 1), TB, 0, stream>>>(
        aggb, W2rb + (size_t)e * HH * OO, hE2b + (size_t)e * NN * HH, W2tb + (size_t)e * HH * OO,
        b2_rel + (size_t)e * OO, NN, HH, OO, 0, 0, 0,
        nullptr, out, nullptr, nullptr, nullptr, wts, e);
  }
}
